// SphericalAttentionStable_44727789420747
// MI455X (gfx1250) — compile-verified, matrix operand order checked by reading, not run on hardware
//
#include <hip/hip_runtime.h>
#include <hip/hip_bf16.h>
#include <stdint.h>

// ---------------------------------------------------------------------------
// SphericalAttentionStable for MI455X (gfx1250, wave32, WMMA bf16->f32 accum,
// TDM double-buffered LDS staging for weights AND K/V, fast polynomial acos,
// split-barrier sync without full-counter fences)
// ---------------------------------------------------------------------------
typedef __bf16 bf16_t;
typedef __attribute__((ext_vector_type(16))) __bf16        v16bf;
typedef __attribute__((ext_vector_type(8)))  __bf16        v8bf;
typedef __attribute__((ext_vector_type(8)))  float         v8f;
typedef __attribute__((ext_vector_type(4)))  unsigned int  v4u;
typedef __attribute__((ext_vector_type(8)))  int           v8i;
typedef __attribute__((ext_vector_type(4)))  int           v4i;

#define D_MODEL   1024
#define N_HEADS   16
#define D_HEAD    64
#define SEQ       2048
#define BATCH     2
#define MTOT      (BATCH * SEQ)      // 4096 token rows
#define BH        (BATCH * N_HEADS)  // 32 (batch, head) pairs
#define INV_SCALE 0.125f             // 1 / sqrt(D_HEAD)
#define EPS_CLAMP 1e-6f
#define EPS_NORM  1e-10f
#define WROW      40                 // LDS row stride, 64B data + 16B pad (80 B)
#define KROW      72                 // LDS row stride, 128B data + 16B pad (144 B)
#define NWAVE     8                  // waves per block (256 threads)

#if defined(__has_builtin)
#if __has_builtin(__builtin_amdgcn_tensor_load_to_lds) && \
    __has_builtin(__builtin_amdgcn_s_wait_tensorcnt)
#define USE_TDM 1
#endif
#endif
#ifndef USE_TDM
#define USE_TDM 0
#endif

#if USE_TDM && defined(__has_builtin)
#if __has_builtin(__builtin_amdgcn_s_barrier_signal) && \
    __has_builtin(__builtin_amdgcn_s_barrier_wait)
#define HAVE_SPLIT_BARRIER 1
#endif
#endif

static __device__ __forceinline__ v8f vzero8() {
    v8f z = {0.f, 0.f, 0.f, 0.f, 0.f, 0.f, 0.f, 0.f};
    return z;
}

// 16-bit WMMA operand fragment: two 16-byte chunks per lane at
// k + (lane>>4)*8 and +16 (CDNA5 ISA §7.12.2 layout).
static __device__ __forceinline__ v16bf load_frag(const bf16_t* p) {
    v8bf lo = *(const v8bf*)(p);
    v8bf hi = *(const v8bf*)(p + 16);
    return __builtin_shufflevector(lo, hi, 0,1,2,3,4,5,6,7,8,9,10,11,12,13,14,15);
}

// Abramowitz-Stegun 4.4.45 minimax acos, |err| <= ~6.7e-5 rad -- far below
// the bf16 operand quantization already accepted.  ~6 VALU + v_sqrt_f32.
static __device__ __forceinline__ float fast_acos(float x) {
    float ax = fabsf(x);
    float p = fmaf(ax, -0.0187293f, 0.0742610f);
    p = fmaf(p, ax, -0.2121144f);
    p = fmaf(p, ax, 1.5707288f);
    float r = sqrtf(1.0f - ax) * p;
    return (x < 0.0f) ? (3.14159265f - r) : r;
}

// ---------------------------------------------------------------------------
// Stage a tile_rows x tile_k bf16 2D slab into LDS via the Tensor Data Mover.
// LDS rows land at lds_row*2 bytes via the TDM pad feature.  Issued by wave 0
// only (TENSORcnt-tracked); cooperative-copy fallback if builtin is absent.
//   pad_code: 3 -> pad every 16 DW (64B rows),  4 -> every 32 DW (128B rows)
// ---------------------------------------------------------------------------
static __device__ __forceinline__ void tdm_stage_2d(bf16_t* slab, const bf16_t* src,
                                                    int tile_k, int tile_rows,
                                                    int g_stride, int lds_row,
                                                    int pad_code, bool leader) {
#if USE_TDM
    if (leader) {
        const unsigned lds_off = (unsigned)(uintptr_t)slab;  // addr[31:0] = LDS offset
        const uint64_t ga = (uint64_t)(uintptr_t)src;
        v4u g0;
        g0[0] = 1u;                                          // count=1 (valid), user mode
        g0[1] = lds_off;                                     // lds_addr
        g0[2] = (unsigned)(ga & 0xFFFFFFFFu);                // global_addr[31:0]
        g0[3] = (unsigned)((ga >> 32) & 0x01FFFFFFu)         // global_addr[56:32]
                | (2u << 30);                                // type = 2 ("image")
        v8i g1;
        g1[0] = (1 << 16)                                    // data_size = 1 (2 bytes)
              | (1 << 20)                                    // pad_enable
              | (pad_code << 22)                             // pad interval
              | (3 << 25);                                   // pad_amount = 4 DW (16 B)
        g1[1] = (int)((g_stride & 0xFFFF) << 16);            // tensor_dim0[15:0]
        g1[2] = (int)(((g_stride >> 16) & 0xFFFF)            // tensor_dim0[31:16]
                | ((1 << 20) & 0xFFFF0000));                 // tensor_dim1[15:0] (large)
        g1[3] = (int)((1 << 4)                               // tensor_dim1[31:16] (large)
                | (tile_k << 16));                           // tile_dim0
        g1[4] = tile_rows;                                   // tile_dim1 (tile_dim2 = 0)
        g1[5] = g_stride;                                    // tensor_dim0_stride lo32
        g1[6] = 0;                                           // stride hi / dim1_stride lo
        g1[7] = 0;
        v4i z4 = {0, 0, 0, 0};
#if defined(__clang_major__) && (__clang_major__ >= 23)
        v8i z8 = {0, 0, 0, 0, 0, 0, 0, 0};
        __builtin_amdgcn_tensor_load_to_lds(g0, g1, z4, z4, z8, 0);
#else
        __builtin_amdgcn_tensor_load_to_lds(g0, g1, z4, z4, 0);
#endif
    }
#else
    (void)leader; (void)pad_code;  // cooperative fallback: 16B chunks
    const int nch = (tile_k >> 3) * tile_rows;
    const int cpr = tile_k >> 3;
    for (int c = threadIdx.x; c < nch; c += blockDim.x) {
        const int row = c / cpr, ch = c - row * cpr;
        *(v8bf*)(slab + row * lds_row + ch * 8) =
            *(const v8bf*)(src + (size_t)row * g_stride + ch * 8);
    }
#endif
}

// Buffer-flip sync.  TDM path: wait only on TENSORcnt then raw split barrier
// (no full loadcnt/dscnt fence -> pipelined A-fragment loads stay in flight
// across the flip).  asm memory clobbers stop the optimizer from hoisting
// ds_loads of the next buffer above the wait.  Fallback keeps __syncthreads.
static __device__ __forceinline__ void sync_stage() {
#if USE_TDM
    __builtin_amdgcn_s_wait_tensorcnt(0);
#endif
#if defined(HAVE_SPLIT_BARRIER)
    asm volatile("" ::: "memory");
    __builtin_amdgcn_s_barrier_signal(-1);
    __builtin_amdgcn_s_barrier_wait(-1);
    asm volatile("" ::: "memory");
#else
    __syncthreads();
#endif
}

// ---------------------------------------------------------------------------
// f32 -> bf16 conversion (grid-stride)
// ---------------------------------------------------------------------------
__global__ void cvt_f32_bf16_kernel(const float* __restrict__ src,
                                    bf16_t* __restrict__ dst, int n) {
    int i = blockIdx.x * blockDim.x + threadIdx.x;
    const int stride = gridDim.x * blockDim.x;
    for (; i < n; i += stride) dst[i] = (bf16_t)src[i];
}

// ---------------------------------------------------------------------------
// Fused projection: Y = H @ W^T + b with per-head unit-sphere normalization.
// Block = 8 waves sharing one head's 64-row weight slab (TDM double-buffered);
// each wave owns a 16-token x 64 output tile; A-fragment loads pipelined.
//   transposed == 0 : out[bh][s][d]   (Q, K)     transposed == 1 : out[bh][d][s] (V)
// ---------------------------------------------------------------------------
__global__ __launch_bounds__(32 * NWAVE)
void proj_sphere_kernel(const bf16_t* __restrict__ Hb,   // [MTOT, D_MODEL]
                        const bf16_t* __restrict__ W,    // [D_MODEL, D_MODEL]
                        const float*  __restrict__ bias, // [D_MODEL]
                        bf16_t* __restrict__ out,
                        int transposed) {
    __shared__ bf16_t Wlds[2][64 * WROW];

    const int wave  = threadIdx.x >> 5;
    const int lane  = threadIdx.x & 31;
    const int n15   = lane & 15;
    const int half  = lane >> 4;
    const int head  = blockIdx.y;
    const int mtile = blockIdx.x * NWAVE + wave;
    const bool leader = (wave == 0);

    const bf16_t* wbase = W + (size_t)head * D_HEAD * D_MODEL;
    const bf16_t* aptr  = Hb + (size_t)(mtile * 16 + n15) * D_MODEL + half * 8;

    tdm_stage_2d(Wlds[0], wbase, 32, 64, D_MODEL, WROW, 3, leader);
    v16bf a_cur = load_frag(aptr);
    sync_stage();

    v8f acc[4] = {vzero8(), vzero8(), vzero8(), vzero8()};

    for (int kk = 0; kk < D_MODEL; kk += 32) {
        const int buf = (kk >> 5) & 1;
        const bool more = (kk + 32) < D_MODEL;
        if (more) tdm_stage_2d(Wlds[buf ^ 1], wbase + kk + 32, 32, 64, D_MODEL, WROW, 3, leader);
        v16bf a_nxt = a_cur;
        if (more) a_nxt = load_frag(aptr + kk + 32);

        const bf16_t* bl = &Wlds[buf][n15 * WROW + half * 8];
        v16bf b0 = load_frag(bl + 0 * 16 * WROW);
        v16bf b1 = load_frag(bl + 1 * 16 * WROW);
        v16bf b2 = load_frag(bl + 2 * 16 * WROW);
        v16bf b3 = load_frag(bl + 3 * 16 * WROW);
        acc[0] = __builtin_amdgcn_wmma_f32_16x16x32_bf16(false, a_cur, false, b0, (short)0, acc[0], false, false);
        acc[1] = __builtin_amdgcn_wmma_f32_16x16x32_bf16(false, a_cur, false, b1, (short)0, acc[1], false, false);
        acc[2] = __builtin_amdgcn_wmma_f32_16x16x32_bf16(false, a_cur, false, b2, (short)0, acc[2], false, false);
        acc[3] = __builtin_amdgcn_wmma_f32_16x16x32_bf16(false, a_cur, false, b3, (short)0, acc[3], false, false);

        a_cur = a_nxt;
        sync_stage();
    }

#pragma unroll
    for (int t = 0; t < 4; ++t)
        acc[t] += bias[head * D_HEAD + t * 16 + n15];

    // row ||y||^2 over the head: in-lane partial (4 cols), butterfly over the
    // 16 lanes of this half (masks 1..8 never cross the half boundary).
    v8f ssv = acc[0] * acc[0] + acc[1] * acc[1] + acc[2] * acc[2] + acc[3] * acc[3];
    v8f invn;
#pragma unroll
    for (int r = 0; r < 8; ++r) {
        float s = ssv[r];
        s += __shfl_xor(s, 1, 32);
        s += __shfl_xor(s, 2, 32);
        s += __shfl_xor(s, 4, 32);
        s += __shfl_xor(s, 8, 32);
        invn[r] = 1.0f / fmaxf(sqrtf(s), EPS_NORM);
    }

    const int batch = (mtile * 16) >> 11;
    const int bh    = batch * N_HEADS + head;
#pragma unroll
    for (int r = 0; r < 8; ++r) {
        const int m = mtile * 16 + r + half * 8;
        const int s = m & (SEQ - 1);
#pragma unroll
        for (int t = 0; t < 4; ++t) {
            const int d = t * 16 + n15;
            const bf16_t v = (bf16_t)(acc[t][r] * invn[r]);
            if (!transposed)
                out[((size_t)bh * SEQ + s) * D_HEAD + d] = v;
            else
                out[((size_t)bh * D_HEAD + d) * SEQ + s] = v;
        }
    }
}

// ---------------------------------------------------------------------------
// Single-pass spherical attention.  scores = -acos(clip(Q.K))/8 in [-pi/8,0]
// -> exp is stable without max tracking (running denominator only).
// K (32x64) and V (64x32) slabs for each 32-key step are TDM double-buffered
// in LDS and shared by all 8 waves; P tiles round-trip per-wave LDS slabs.
// ---------------------------------------------------------------------------
__global__ __launch_bounds__(32 * NWAVE)
void spherical_attn_kernel(const bf16_t* __restrict__ Qb,  // [BH, SEQ, D_HEAD]
                           const bf16_t* __restrict__ Kb,  // [BH, SEQ, D_HEAD]
                           const bf16_t* __restrict__ Vt,  // [BH, D_HEAD, SEQ]
                           bf16_t* __restrict__ Ob) {      // [MTOT, D_MODEL]
    __shared__ bf16_t Klds[2][32 * KROW];        // 32 keys x 64 d, 144B rows
    __shared__ bf16_t Vlds[2][64 * WROW];        // 64 d    x 32 k, 80B rows
    __shared__ bf16_t Plds[NWAVE][16 * WROW];    // per-wave P tile

    const int wave  = threadIdx.x >> 5;
    const int lane  = threadIdx.x & 31;
    const int n15   = lane & 15;
    const int half  = lane >> 4;
    const int bh    = blockIdx.x;
    const int qtile = blockIdx.y * NWAVE + wave;
    const int qrow  = qtile * 16 + n15;
    const bool leader = (wave == 0);

    const bf16_t* Kbh = Kb + (size_t)bh * SEQ * D_HEAD;
    const bf16_t* Vbh = Vt + (size_t)bh * D_HEAD * SEQ;

    const bf16_t* qbase = Qb + ((size_t)bh * SEQ + qrow) * D_HEAD + half * 8;
    const v16bf a0 = load_frag(qbase);            // d = 0..31
    const v16bf a1 = load_frag(qbase + 32);       // d = 32..63

    tdm_stage_2d(Klds[0], Kbh, 64, 32, D_HEAD, KROW, 4, leader);
    tdm_stage_2d(Vlds[0], Vbh, 32, 64, SEQ, WROW, 3, leader);
    sync_stage();

    v8f oacc[4] = {vzero8(), vzero8(), vzero8(), vzero8()};
    v8f rsum = vzero8();

    for (int j = 0; j < SEQ; j += 32) {
        const int buf = (j >> 5) & 1;
        const bool more = (j + 32) < SEQ;
        if (more) {
            tdm_stage_2d(Klds[buf ^ 1], Kbh + (size_t)(j + 32) * D_HEAD, 64, 32, D_HEAD, KROW, 4, leader);
            tdm_stage_2d(Vlds[buf ^ 1], Vbh + j + 32, 32, 64, SEQ, WROW, 3, leader);
        }

        // -------- B fragments from the shared slabs
        const bf16_t* kl = &Klds[buf][n15 * KROW + half * 8];
        v16bf k00 = load_frag(kl);                 // keys j..j+15,  d 0..31
        v16bf k01 = load_frag(kl + 32);            //                d 32..63
        v16bf k10 = load_frag(kl + 16 * KROW);     // keys j+16..j+31
        v16bf k11 = load_frag(kl + 16 * KROW + 32);
        const bf16_t* vl = &Vlds[buf][n15 * WROW + half * 8];
        v16bf vb0 = load_frag(vl + 0 * 16 * WROW); // d rows n15 + 16t
        v16bf vb1 = load_frag(vl + 1 * 16 * WROW);
        v16bf vb2 = load_frag(vl + 2 * 16 * WROW);
        v16bf vb3 = load_frag(vl + 3 * 16 * WROW);

        // -------- scores (two 16x16 tiles over K-dim 64)
        v8f s0 = vzero8(), s1 = vzero8();
        s0 = __builtin_amdgcn_wmma_f32_16x16x32_bf16(false, a0, false, k00, (short)0, s0, false, false);
        s0 = __builtin_amdgcn_wmma_f32_16x16x32_bf16(false, a1, false, k01, (short)0, s0, false, false);
        s1 = __builtin_amdgcn_wmma_f32_16x16x32_bf16(false, a0, false, k10, (short)0, s1, false, false);
        s1 = __builtin_amdgcn_wmma_f32_16x16x32_bf16(false, a1, false, k11, (short)0, s1, false, false);

        // -------- p = exp(-acos(clip(cos))/8); stage P tile into LDS
#pragma unroll
        for (int r = 0; r < 8; ++r) {
            float c0 = fminf(fmaxf(s0[r], -1.0f + EPS_CLAMP), 1.0f - EPS_CLAMP);
            float c1 = fminf(fmaxf(s1[r], -1.0f + EPS_CLAMP), 1.0f - EPS_CLAMP);
            float p0 = __expf(-fast_acos(c0) * INV_SCALE);
            float p1 = __expf(-fast_acos(c1) * INV_SCALE);
            rsum[r] += p0 + p1;
            Plds[wave][(r + half * 8) * WROW + n15]      = (bf16_t)p0;
            Plds[wave][(r + half * 8) * WROW + 16 + n15] = (bf16_t)p1;
        }
        __threadfence_block();   // order P stores before the A-fragment reload

        const v16bf pa = load_frag(&Plds[wave][n15 * WROW + half * 8]);

        // -------- O += P @ V
        oacc[0] = __builtin_amdgcn_wmma_f32_16x16x32_bf16(false, pa, false, vb0, (short)0, oacc[0], false, false);
        oacc[1] = __builtin_amdgcn_wmma_f32_16x16x32_bf16(false, pa, false, vb1, (short)0, oacc[1], false, false);
        oacc[2] = __builtin_amdgcn_wmma_f32_16x16x32_bf16(false, pa, false, vb2, (short)0, oacc[2], false, false);
        oacc[3] = __builtin_amdgcn_wmma_f32_16x16x32_bf16(false, pa, false, vb3, (short)0, oacc[3], false, false);

        sync_stage();            // K/V buffer flip; also orders next P stores
    }

    // softmax denominator across the 16 lanes of each half
    v8f invl;
#pragma unroll
    for (int r = 0; r < 8; ++r) {
        float s = rsum[r];
        s += __shfl_xor(s, 1, 32);
        s += __shfl_xor(s, 2, 32);
        s += __shfl_xor(s, 4, 32);
        s += __shfl_xor(s, 8, 32);
        invl[r] = 1.0f / s;
    }

    const int batch = bh >> 4;
    const int head  = bh & 15;
#pragma unroll
    for (int r = 0; r < 8; ++r) {
        const int s    = qtile * 16 + r + half * 8;
        const int mrow = batch * SEQ + s;
#pragma unroll
        for (int t = 0; t < 4; ++t)
            Ob[(size_t)mrow * D_MODEL + head * D_HEAD + t * 16 + n15] =
                (bf16_t)(oacc[t][r] * invl[r]);
    }
}

// ---------------------------------------------------------------------------
// Output projection: out = Ob @ Wo^T + b_o  (f32 stores), same TDM-staged
// weight-slab structure as the projection kernel.
// ---------------------------------------------------------------------------
__global__ __launch_bounds__(32 * NWAVE)
void out_proj_kernel(const bf16_t* __restrict__ Ab,   // [MTOT, D_MODEL]
                     const bf16_t* __restrict__ W,    // [D_MODEL, D_MODEL]
                     const float*  __restrict__ bias, // [D_MODEL]
                     float* __restrict__ out) {       // [MTOT, D_MODEL]
    __shared__ bf16_t Wlds[2][64 * WROW];

    const int wave  = threadIdx.x >> 5;
    const int lane  = threadIdx.x & 31;
    const int n15   = lane & 15;
    const int half  = lane >> 4;
    const int ncol  = blockIdx.y;
    const int mtile = blockIdx.x * NWAVE + wave;
    const bool leader = (wave == 0);

    const bf16_t* wbase = W + (size_t)ncol * D_HEAD * D_MODEL;
    const bf16_t* aptr  = Ab + (size_t)(mtile * 16 + n15) * D_MODEL + half * 8;

    tdm_stage_2d(Wlds[0], wbase, 32, 64, D_MODEL, WROW, 3, leader);
    v16bf a_cur = load_frag(aptr);
    sync_stage();

    v8f acc[4] = {vzero8(), vzero8(), vzero8(), vzero8()};

    for (int kk = 0; kk < D_MODEL; kk += 32) {
        const int buf = (kk >> 5) & 1;
        const bool more = (kk + 32) < D_MODEL;
        if (more) tdm_stage_2d(Wlds[buf ^ 1], wbase + kk + 32, 32, 64, D_MODEL, WROW, 3, leader);
        v16bf a_nxt = a_cur;
        if (more) a_nxt = load_frag(aptr + kk + 32);

        const bf16_t* bl = &Wlds[buf][n15 * WROW + half * 8];
        v16bf b0 = load_frag(bl + 0 * 16 * WROW);
        v16bf b1 = load_frag(bl + 1 * 16 * WROW);
        v16bf b2 = load_frag(bl + 2 * 16 * WROW);
        v16bf b3 = load_frag(bl + 3 * 16 * WROW);
        acc[0] = __builtin_amdgcn_wmma_f32_16x16x32_bf16(false, a_cur, false, b0, (short)0, acc[0], false, false);
        acc[1] = __builtin_amdgcn_wmma_f32_16x16x32_bf16(false, a_cur, false, b1, (short)0, acc[1], false, false);
        acc[2] = __builtin_amdgcn_wmma_f32_16x16x32_bf16(false, a_cur, false, b2, (short)0, acc[2], false, false);
        acc[3] = __builtin_amdgcn_wmma_f32_16x16x32_bf16(false, a_cur, false, b3, (short)0, acc[3], false, false);

        a_cur = a_nxt;
        sync_stage();
    }

#pragma unroll
    for (int t = 0; t < 4; ++t)
        acc[t] += bias[ncol * 64 + t * 16 + n15];

#pragma unroll
    for (int r = 0; r < 8; ++r) {
        const int m = mtile * 16 + r + half * 8;
#pragma unroll
        for (int t = 0; t < 4; ++t)
            out[(size_t)m * D_MODEL + ncol * 64 + t * 16 + n15] = acc[t][r];
    }
}

// ---------------------------------------------------------------------------
// Host-side orchestration
// ---------------------------------------------------------------------------
extern "C" void kernel_launch(void* const* d_in, const int* in_sizes, int n_in,
                              void* d_out, int out_size, void* d_ws, size_t ws_size,
                              hipStream_t stream) {
    (void)in_sizes; (void)n_in; (void)out_size; (void)ws_size;

    const float* H  = (const float*)d_in[0];
    const float* Wq = (const float*)d_in[1];
    const float* bq = (const float*)d_in[2];
    const float* Wk = (const float*)d_in[3];
    const float* bk = (const float*)d_in[4];
    const float* Wv = (const float*)d_in[5];
    const float* bv = (const float*)d_in[6];
    const float* Wo = (const float*)d_in[7];
    const float* bo = (const float*)d_in[8];
    float* out = (float*)d_out;

    char* ws = (char*)d_ws;
    const size_t MB = size_t(1) << 20;
    bf16_t* Hb  = (bf16_t*)(ws + 0 * MB);   // [4096,1024]   8 MiB
    bf16_t* Wqb = (bf16_t*)(ws + 8 * MB);   // [1024,1024]   2 MiB each
    bf16_t* Wkb = (bf16_t*)(ws + 10 * MB);
    bf16_t* Wvb = (bf16_t*)(ws + 12 * MB);
    bf16_t* Wob = (bf16_t*)(ws + 14 * MB);
    bf16_t* Qb  = (bf16_t*)(ws + 16 * MB);  // [32,2048,64]  8 MiB
    bf16_t* Kb  = (bf16_t*)(ws + 24 * MB);  // [32,2048,64]  8 MiB
    bf16_t* Vt  = (bf16_t*)(ws + 32 * MB);  // [32,64,2048]  8 MiB
    bf16_t* Ob  = (bf16_t*)(ws + 40 * MB);  // [4096,1024]   8 MiB

    // 1) precision conversion
    cvt_f32_bf16_kernel<<<2048, 256, 0, stream>>>(H,  Hb,  MTOT * D_MODEL);
    cvt_f32_bf16_kernel<<<1024, 256, 0, stream>>>(Wq, Wqb, D_MODEL * D_MODEL);
    cvt_f32_bf16_kernel<<<1024, 256, 0, stream>>>(Wk, Wkb, D_MODEL * D_MODEL);
    cvt_f32_bf16_kernel<<<1024, 256, 0, stream>>>(Wv, Wvb, D_MODEL * D_MODEL);
    cvt_f32_bf16_kernel<<<1024, 256, 0, stream>>>(Wo, Wob, D_MODEL * D_MODEL);

    // 2) fused projection + bias + unit-sphere normalization
    dim3 gproj(MTOT / (16 * NWAVE), N_HEADS);   // 8 m-tiles share a weight slab
    proj_sphere_kernel<<<gproj, 32 * NWAVE, 0, stream>>>(Hb, Wqb, bq, Qb, 0);
    proj_sphere_kernel<<<gproj, 32 * NWAVE, 0, stream>>>(Hb, Wkb, bk, Kb, 0);
    proj_sphere_kernel<<<gproj, 32 * NWAVE, 0, stream>>>(Hb, Wvb, bv, Vt, 1);

    // 3) single-pass spherical attention (8 q-tiles share each K/V slab)
    dim3 gattn(BH, SEQ / (16 * NWAVE));
    spherical_attn_kernel<<<gattn, 32 * NWAVE, 0, stream>>>(Qb, Kb, Vt, Ob);

    // 4) output projection (f32 result)
    dim3 gout(MTOT / (16 * NWAVE), D_MODEL / 64);
    out_proj_kernel<<<gout, 32 * NWAVE, 0, stream>>>(Ob, Wob, bo, out);
}
